// WindowAttention3D_65927747994175
// MI455X (gfx1250) — matrix operand order repaired and checked by reading, not compiled, not run on hardware
//
#include <hip/hip_runtime.h>
#include <hip/hip_bf16.h>

// ---------------------------------------------------------------------------
// Fused Swin-3D window attention for MI455X (gfx1250, wave32, WMMA).
// Memory-bound workload (~67 GFLOP vs ~400MB DRAM traffic @ 23.3 TB/s), so:
//  * never materialize the 1.26GB attention matrix -> flash-style online softmax
//  * f16 storage for Q/K/V/O intermediates, f32 WMMA accumulation
//  * v_wmma_f32_16x16x32_f16 for all four matmuls (hd = 32 = K of one WMMA)
// Workspace use: ~218 MB.
// ---------------------------------------------------------------------------

#define B__   512
#define NTOK  392
#define CDIM  128
#define NH    4
#define HD    32
#define NW    128
#define NPAD  416              // 13 chunks of 32 keys / 26 tiles of 16
#define MTOT  (B__ * NTOK)     // 200704 rows, divisible by 16

typedef __attribute__((ext_vector_type(16))) _Float16 v16h;
typedef __attribute__((ext_vector_type(8)))  _Float16 v8h;
typedef __attribute__((ext_vector_type(8)))  float    v8f;

union V16 { v16h v; v8h h2[2]; _Float16 e[16]; };

__device__ __forceinline__ v8f wmma_f16(v16h a, v16h b, v8f c) {
    // v_wmma_f32_16x16x32_f16 : D = A(16x32) * B(32x16) + C(16x16 f32)
    return __builtin_amdgcn_wmma_f32_16x16x32_f16(
        /*neg_a=*/false, a, /*neg_b=*/false, b,
        /*c_mod=*/(short)0, c, /*reuse_a=*/false, /*reuse_b=*/false);
}

// ---------------------------------------------------------------------------
// prep: f32->f16 weight conversion, bias[h][n][m] gather precompute,
// zero padding of Q/K rows [392,416) and V^T columns [392,416).
// ---------------------------------------------------------------------------
__global__ void wa3d_prep(const float* __restrict__ qkv_w,
                          const float* __restrict__ proj_w,
                          const float* __restrict__ bias_table,
                          const int*   __restrict__ rel_index,
                          _Float16* __restrict__ w_qkv_h,
                          _Float16* __restrict__ w_proj_h,
                          float*    __restrict__ bias_full,
                          _Float16* __restrict__ q_ws,
                          _Float16* __restrict__ k_ws,
                          _Float16* __restrict__ vT_ws)
{
    const long QKVW  = 3L * CDIM * CDIM;          // 49152
    const long PROJW = (long)CDIM * CDIM;         // 16384
    const long BIASF = (long)NH * NTOK * NTOK;    // 614656
    const long PADQ  = 2048L * (NPAD - NTOK) * HD;// 1572864 per buffer
    const long total = QKVW + PROJW + BIASF + 3L * PADQ;

    for (long i = (long)blockIdx.x * blockDim.x + threadIdx.x; i < total;
         i += (long)gridDim.x * blockDim.x) {
        long t = i;
        if (t < QKVW)  { w_qkv_h[t]  = (_Float16)qkv_w[t];  continue; }
        t -= QKVW;
        if (t < PROJW) { w_proj_h[t] = (_Float16)proj_w[t]; continue; }
        t -= PROJW;
        if (t < BIASF) {                       // bias_full[(h*N+n)*N+m]
            long h  = t / ((long)NTOK * NTOK);
            long nm = t % ((long)NTOK * NTOK);
            bias_full[t] = bias_table[(long)rel_index[nm] * NH + h];
            continue;
        }
        t -= BIASF;
        if (t < PADQ) {                        // Q row pad
            long bh  = t / ((NPAD - NTOK) * HD);
            long rd  = t % ((NPAD - NTOK) * HD);
            long row = NTOK + rd / HD, d = rd % HD;
            q_ws[bh * NPAD * HD + row * HD + d] = (_Float16)0.f;
            continue;
        }
        t -= PADQ;
        if (t < PADQ) {                        // K row pad
            long bh  = t / ((NPAD - NTOK) * HD);
            long rd  = t % ((NPAD - NTOK) * HD);
            long row = NTOK + rd / HD, d = rd % HD;
            k_ws[bh * NPAD * HD + row * HD + d] = (_Float16)0.f;
            continue;
        }
        t -= PADQ;
        {                                      // V^T column pad
            long bh = t / ((long)HD * (NPAD - NTOK));
            long dn = t % ((long)HD * (NPAD - NTOK));
            long d  = dn / (NPAD - NTOK);
            long n  = NTOK + dn % (NPAD - NTOK);
            vT_ws[bh * HD * NPAD + d * NPAD + n] = (_Float16)0.f;
        }
    }
}

// ---------------------------------------------------------------------------
// QKV GEMM: (200704 x 128) x (128 x 384). One wave per 16x16 tile, 4 WMMAs.
// Epilogue scatters into attention-friendly layouts:
//   Q (scaled, row-major Nx32), K (row-major Nx32), V^T (32 x NPAD).
// ---------------------------------------------------------------------------
__global__ void wa3d_qkv_gemm(const float* __restrict__ x,
                              const _Float16* __restrict__ w_qkv_h,
                              _Float16* __restrict__ q_ws,
                              _Float16* __restrict__ k_ws,
                              _Float16* __restrict__ vT_ws)
{
    const int lane = threadIdx.x & 31, wv = threadIdx.x >> 5;
    const int half = lane >> 4, l16 = lane & 15;
    const int mtile = blockIdx.x;
    const int nbase = blockIdx.y * 64 + wv * 16;
    const int arow  = mtile * 16 + l16;
    const float* xr = x + (size_t)arow * CDIM;
    const int col   = nbase + l16;
    const _Float16* wr = w_qkv_h + (size_t)col * CDIM;

    v8f acc = {};
    #pragma unroll
    for (int k0 = 0; k0 < CDIM; k0 += 32) {
        V16 a;                                   // A layout: K 0-15/16-31 contiguous
        const int ka = k0 + 16 * half, kb = k0 + 16 * half + 8;
        #pragma unroll
        for (int i = 0; i < 8; ++i) {
            a.e[i]     = (_Float16)xr[ka + i];
            a.e[8 + i] = (_Float16)xr[kb + i];
        }
        v16h b = *(const v16h*)(wr + k0 + 16 * half);   // B: 16 contiguous K
        acc = wmma_f16(a.v, b, acc);
    }

    const float scale = 0.17677669529663687f;    // 32^-0.5
    #pragma unroll
    for (int r = 0; r < 8; ++r) {
        const int orow = mtile * 16 + r + 8 * half;     // D layout row
        const int b_ = orow / NTOK, n = orow % NTOK;
        const float v = acc[r];
        const int sec = col >> 7, wi = col & 127;
        const int h = wi >> 5, d = wi & 31;
        const size_t bh = (size_t)b_ * NH + h;
        if (sec == 0)      q_ws [bh * NPAD * HD + (size_t)n * HD + d] = (_Float16)(v * scale);
        else if (sec == 1) k_ws [bh * NPAD * HD + (size_t)n * HD + d] = (_Float16)v;
        else               vT_ws[bh * HD * NPAD + (size_t)d * NPAD + n] = (_Float16)v;
    }
}

// ---------------------------------------------------------------------------
// Fused attention: one wave per (b,h, 16-query-row tile). Flash-style online
// softmax over 13 chunks of 32 keys; 4 WMMAs per chunk. P transposed
// D->A layout through a 1KB LDS staging tile.
// ---------------------------------------------------------------------------
__global__ void wa3d_attn(const float* __restrict__ mask,
                          const float* __restrict__ bias_full,
                          const _Float16* __restrict__ q_ws,
                          const _Float16* __restrict__ k_ws,
                          const _Float16* __restrict__ vT_ws,
                          _Float16* __restrict__ o_ws)
{
    __shared__ __align__(32) _Float16 Plds[16 * 32];

    const int lane = threadIdx.x & 31;
    const int half = lane >> 4, l16 = lane & 15;
    const int qt = blockIdx.x;                  // 0..24 query tiles
    const int bh = blockIdx.y;                  // b_*4 + h
    const int b_ = bh >> 2, h = bh & 3;
    const int w  = b_ & (NW - 1);               // window index for mask

    const _Float16* qp = q_ws  + (size_t)bh * NPAD * HD;
    const _Float16* kp = k_ws  + (size_t)bh * NPAD * HD;
    const _Float16* vp = vT_ws + (size_t)bh * HD * NPAD;
    const float* bp = bias_full + (size_t)h * NTOK * NTOK;
    const float* mp = mask      + (size_t)w * NTOK * NTOK;

    // Q A-tile (16 rows x K=32), zero-padded rows beyond 392
    V16 aq;
    {
        const _Float16* qr = qp + (size_t)(qt * 16 + l16) * HD;
        aq.h2[0] = *(const v8h*)(qr + 16 * half);
        aq.h2[1] = *(const v8h*)(qr + 16 * half + 8);
    }

    float mrow[8], lrow[8];
    v8f o0 = {}, o1 = {};
    #pragma unroll
    for (int r = 0; r < 8; ++r) { mrow[r] = -1e30f; lrow[r] = 0.f; }

    for (int kt = 0; kt < 13; ++kt) {
        const int k0 = kt * 32;
        // K B-tiles: column j = key index, 16 contiguous K per lane
        v16h b0 = *(const v16h*)(kp + (size_t)(k0 + l16)      * HD + 16 * half);
        v16h b1 = *(const v16h*)(kp + (size_t)(k0 + 16 + l16) * HD + 16 * half);
        if (kt < 12)
            __builtin_prefetch(vp + (size_t)l16 * NPAD + k0 + 32, 0, 0);

        v8f zc = {};
        v8f s0 = wmma_f16(aq.v, b0, zc);
        v8f s1 = wmma_f16(aq.v, b1, zc);

        const int m0 = k0 + l16, m1 = m0 + 16;
        #pragma unroll
        for (int r = 0; r < 8; ++r) {
            const int n = qt * 16 + r + 8 * half;
            float s0v = s0[r], s1v = s1[r];
            if (n < NTOK && m0 < NTOK)
                s0v += bp[(size_t)n * NTOK + m0] + mp[(size_t)n * NTOK + m0];
            if (m0 >= NTOK) s0v = -1e30f;
            if (n < NTOK && m1 < NTOK)
                s1v += bp[(size_t)n * NTOK + m1] + mp[(size_t)n * NTOK + m1];
            if (m1 >= NTOK) s1v = -1e30f;

            // row reductions across the 16 lanes holding this row (wave32)
            float tmax = fmaxf(s0v, s1v);
            #pragma unroll
            for (int sh = 1; sh < 16; sh <<= 1)
                tmax = fmaxf(tmax, __shfl_xor(tmax, sh, 32));
            const float mnew = fmaxf(mrow[r], tmax);
            const float corr = __expf(mrow[r] - mnew);
            const float p0 = __expf(s0v - mnew);
            const float p1 = __expf(s1v - mnew);
            float rs = p0 + p1;
            #pragma unroll
            for (int sh = 1; sh < 16; sh <<= 1)
                rs += __shfl_xor(rs, sh, 32);
            lrow[r] = lrow[r] * corr + rs;
            mrow[r] = mnew;
            o0[r] *= corr;
            o1[r] *= corr;

            const int prow = r + 8 * half;       // stage P for D->A transpose
            Plds[prow * 32 + l16]      = (_Float16)p0;
            Plds[prow * 32 + 16 + l16] = (_Float16)p1;
        }
        __syncthreads();

        // P as A-operand (16x32 f16)
        V16 pa;
        pa.h2[0] = *(const v8h*)(&Plds[l16 * 32 + 16 * half]);
        pa.h2[1] = *(const v8h*)(&Plds[l16 * 32 + 16 * half + 8]);

        // V B-tiles from V^T: column = output dim d, 16 contiguous keys
        v16h vb0 = *(const v16h*)(vp + (size_t)l16        * NPAD + k0 + 16 * half);
        v16h vb1 = *(const v16h*)(vp + (size_t)(16 + l16) * NPAD + k0 + 16 * half);
        o0 = wmma_f16(pa.v, vb0, o0);
        o1 = wmma_f16(pa.v, vb1, o1);
        __syncthreads();
    }

    #pragma unroll
    for (int r = 0; r < 8; ++r) {
        const int n = qt * 16 + r + 8 * half;
        if (n < NTOK) {
            const float inv = 1.f / lrow[r];
            const size_t base = ((size_t)b_ * NTOK + n) * CDIM + h * HD;
            o_ws[base + l16]      = (_Float16)(o0[r] * inv);
            o_ws[base + 16 + l16] = (_Float16)(o1[r] * inv);
        }
    }
}

// ---------------------------------------------------------------------------
// Output projection: (200704 x 128) x (128 x 128) + bias, f32 out.
// ---------------------------------------------------------------------------
__global__ void wa3d_proj_gemm(const _Float16* __restrict__ o_ws,
                               const _Float16* __restrict__ w_proj_h,
                               const float* __restrict__ proj_b,
                               float* __restrict__ out)
{
    const int lane = threadIdx.x & 31, wv = threadIdx.x >> 5;
    const int half = lane >> 4, l16 = lane & 15;
    const int mtile = blockIdx.x;
    const int nbase = blockIdx.y * 64 + wv * 16;
    const _Float16* ar = o_ws + (size_t)(mtile * 16 + l16) * CDIM;
    const int col = nbase + l16;
    const _Float16* wr = w_proj_h + (size_t)col * CDIM;

    v8f acc = {};
    #pragma unroll
    for (int k0 = 0; k0 < CDIM; k0 += 32) {
        V16 a;
        a.h2[0] = *(const v8h*)(ar + k0 + 16 * half);
        a.h2[1] = *(const v8h*)(ar + k0 + 16 * half + 8);
        v16h b = *(const v16h*)(wr + k0 + 16 * half);
        acc = wmma_f16(a.v, b, acc);
    }
    const float bias = proj_b[col];
    #pragma unroll
    for (int r = 0; r < 8; ++r) {
        const int orow = mtile * 16 + r + 8 * half;
        out[(size_t)orow * CDIM + col] = acc[r] + bias;
    }
}

// ---------------------------------------------------------------------------
extern "C" void kernel_launch(void* const* d_in, const int* in_sizes, int n_in,
                              void* d_out, int out_size, void* d_ws, size_t ws_size,
                              hipStream_t stream)
{
    const float* x          = (const float*)d_in[0];
    const float* mask       = (const float*)d_in[1];
    const float* qkv_w      = (const float*)d_in[2];
    const float* proj_w     = (const float*)d_in[3];
    const float* proj_b     = (const float*)d_in[4];
    const float* bias_table = (const float*)d_in[5];
    const int*   rel_index  = (const int*)d_in[6];
    float* out = (float*)d_out;

    // workspace carve-out (~218 MB)
    char* ws = (char*)d_ws;
    size_t off = 0;
    auto take = [&](size_t bytes) {
        void* p = ws + off;
        off = (off + bytes + 255) & ~(size_t)255;
        return p;
    };
    _Float16* w_qkv_h  = (_Float16*)take((size_t)3 * CDIM * CDIM * 2);
    _Float16* w_proj_h = (_Float16*)take((size_t)CDIM * CDIM * 2);
    float*    bias_full= (float*)   take((size_t)NH * NTOK * NTOK * 4);
    _Float16* q_ws     = (_Float16*)take((size_t)B__ * NH * NPAD * HD * 2);
    _Float16* k_ws     = (_Float16*)take((size_t)B__ * NH * NPAD * HD * 2);
    _Float16* vT_ws    = (_Float16*)take((size_t)B__ * NH * HD * NPAD * 2);
    _Float16* o_ws     = (_Float16*)take((size_t)MTOT * CDIM * 2);
    (void)ws_size; (void)in_sizes; (void)n_in; (void)out_size;

    wa3d_prep<<<dim3(2048), dim3(256), 0, stream>>>(
        qkv_w, proj_w, bias_table, rel_index,
        w_qkv_h, w_proj_h, bias_full, q_ws, k_ws, vT_ws);

    wa3d_qkv_gemm<<<dim3(MTOT / 16, 6), dim3(128), 0, stream>>>(
        x, w_qkv_h, q_ws, k_ws, vT_ws);

    wa3d_attn<<<dim3(25, (unsigned)B__ * NH), dim3(32), 0, stream>>>(
        mask, bias_full, q_ws, k_ws, vT_ws, o_ws);

    wa3d_proj_gemm<<<dim3(MTOT / 16, 2), dim3(128), 0, stream>>>(
        o_ws, w_proj_h, proj_b, out);
}
